// Transformer_82927228551541
// MI455X (gfx1250) — matrix operand order repaired and checked by reading, not compiled, not run on hardware
//
#include <hip/hip_runtime.h>
#include <hip/hip_bf16.h>
#include <math.h>

typedef __attribute__((ext_vector_type(16))) _Float16 v16h;
typedef __attribute__((ext_vector_type(8)))  float    v8f;

#define DIMC   256
#define HEADS  8
#define HDIM   32
#define MLPC   1024
#define JTOK   17
#define NPAD   288     // max tokens (273) padded to multiple of 16
#define BATCH  64
#define SCALE  0.0625f // 256^-0.5
#define NEGINF (-1.0e30f)

// ---------------- half-wave (16-lane) reductions: rows live in 16-lane groups ----------------
__device__ inline float hredMax(float x) {
    x = fmaxf(x, __shfl_xor(x, 1, 32));
    x = fmaxf(x, __shfl_xor(x, 2, 32));
    x = fmaxf(x, __shfl_xor(x, 4, 32));
    x = fmaxf(x, __shfl_xor(x, 8, 32));
    return x;
}
__device__ inline float hredSum(float x) {
    x += __shfl_xor(x, 1, 32);
    x += __shfl_xor(x, 2, 32);
    x += __shfl_xor(x, 4, 32);
    x += __shfl_xor(x, 8, 32);
    return x;
}

// ---------------- weight convert f32 [L][K][N] -> f16 transposed [L][N][K] ----------------
__global__ void cvt_transpose(const float* __restrict__ src, _Float16* __restrict__ dst,
                              int K, int N, int total) {
    for (int i = blockIdx.x * blockDim.x + threadIdx.x; i < total; i += gridDim.x * blockDim.x) {
        int l = i / (K * N);
        int r = i - l * K * N;
        int k = r / N;
        int n = r - k * N;
        dst[(size_t)l * K * N + (size_t)n * K + k] = (_Float16)src[i];
    }
}

// ---------------- init x (pad rows zero) and broadcast pos ----------------
__global__ void init_x(const float* __restrict__ xin, float* __restrict__ xA, int total) {
    for (int i = blockIdx.x * blockDim.x + threadIdx.x; i < total; i += gridDim.x * blockDim.x) {
        int b = i / (NPAD * DIMC);
        int r = i - b * NPAD * DIMC;
        int row = r / DIMC;
        int d = r - row * DIMC;
        xA[i] = (row < JTOK + 256) ? xin[((size_t)b * (JTOK + 256) + row) * DIMC + d] : 0.0f;
    }
}
__global__ void init_pos(const float* __restrict__ pin, float* __restrict__ posA, int total) {
    for (int i = blockIdx.x * blockDim.x + threadIdx.x; i < total; i += gridDim.x * blockDim.x)
        posA[i] = pin[i % (256 * DIMC)];
}

// ---------------- x[:, J:J+P] += pos ----------------
__global__ void add_pos_kernel(float* __restrict__ x, const float* __restrict__ pos, int P) {
    int b = blockIdx.x / P;
    int p = blockIdx.x - b * P;
    int tid = threadIdx.x;
    x[((size_t)b * NPAD + JTOK + p) * DIMC + tid] += pos[((size_t)b * 256 + p) * DIMC + tid];
}

// ---------------- LayerNorm(256) -> f16 ; zero pad rows ----------------
__global__ void ln_kernel(const float* __restrict__ x, const float* __restrict__ g,
                          const float* __restrict__ beta, _Float16* __restrict__ out, int Nin) {
    int row = blockIdx.x;
    int lr = row % NPAD;
    int tid = threadIdx.x;
    size_t base = (size_t)row * DIMC;
    __shared__ float red[256];
    if (lr >= Nin) { out[base + tid] = (_Float16)0.0f; return; }
    float xv = x[base + tid];
    red[tid] = xv; __syncthreads();
    for (int s = 128; s > 0; s >>= 1) { if (tid < s) red[tid] += red[tid + s]; __syncthreads(); }
    float mean = red[0] * (1.0f / DIMC); __syncthreads();
    float d = xv - mean;
    red[tid] = d * d; __syncthreads();
    for (int s = 128; s > 0; s >>= 1) { if (tid < s) red[tid] += red[tid + s]; __syncthreads(); }
    float var = red[0] * (1.0f / DIMC);
    float r = rsqrtf(var + 1e-5f);
    out[base + tid] = (_Float16)(d * r * g[tid] + beta[tid]);
}

// ---------------- WMMA GEMM, 16x64 tile per wave (A-frag reused across 4 N-tiles) ----------------
// A[f16, rows batch*NPAD x K] @ WT[f16, N x K]
// mode 0: store f16 (no bias)    mode 1: f32 = C + bias + residual    mode 2: f16 = GELU(C + bias)
__global__ void gemm_wmma(const _Float16* __restrict__ A, const _Float16* __restrict__ WT,
                          const float* __restrict__ bias, const float* __restrict__ resid,
                          void* __restrict__ out, int Kdim, int Nc, int tpb, int TM, int mode) {
    int l = threadIdx.x & 31;
    int w = threadIdx.x >> 5;
    int tm = blockIdx.y * 8 + w;
    if (tm >= TM) return;
    int batch = tm / tpb;
    int lt = tm - batch * tpb;
    int rowBase = batch * NPAD + lt * 16;
    int col0 = blockIdx.x * 64;
    int hi = l >> 4, ln = l & 15;
    const _Float16* ap = A + (size_t)(rowBase + ln) * Kdim;
    const _Float16* bp = WT + (size_t)(col0 + ln) * Kdim + hi * 16;
    v8f c0 = {}, c1 = {}, c2 = {}, c3 = {};
    const size_t bstride = (size_t)16 * Kdim;
    for (int k0 = 0; k0 < Kdim; k0 += 32) {
        v16h a, b0, b1, b2, b3;
        const _Float16* pa = ap + k0 + hi * 16;
        #pragma unroll
        for (int e = 0; e < 8; ++e) { a[e] = pa[e]; a[8 + e] = pa[8 + e]; }
        const _Float16* pb = bp + k0;
        #pragma unroll
        for (int e = 0; e < 16; ++e) {
            b0[e] = pb[e];
            b1[e] = pb[bstride + e];
            b2[e] = pb[2 * bstride + e];
            b3[e] = pb[3 * bstride + e];
        }
        c0 = __builtin_amdgcn_wmma_f32_16x16x32_f16(false, a, false, b0, (short)0, c0, false, false);
        c1 = __builtin_amdgcn_wmma_f32_16x16x32_f16(false, a, false, b1, (short)0, c1, false, false);
        c2 = __builtin_amdgcn_wmma_f32_16x16x32_f16(false, a, false, b2, (short)0, c2, false, false);
        c3 = __builtin_amdgcn_wmma_f32_16x16x32_f16(false, a, false, b3, (short)0, c3, false, false);
    }
    int rb = hi * 8;
    v8f acc[4] = {c0, c1, c2, c3};
    #pragma unroll
    for (int t = 0; t < 4; ++t) {
        int col = col0 + t * 16 + ln;
        if (mode == 0) {
            _Float16* o = (_Float16*)out;
            #pragma unroll
            for (int v = 0; v < 8; ++v)
                o[(size_t)(rowBase + rb + v) * Nc + col] = (_Float16)acc[t][v];
        } else if (mode == 1) {
            float* o = (float*)out;
            float bb = bias[col];
            #pragma unroll
            for (int v = 0; v < 8; ++v) {
                size_t idx = (size_t)(rowBase + rb + v) * Nc + col;
                o[idx] = acc[t][v] + bb + resid[idx];
            }
        } else {
            _Float16* o = (_Float16*)out;
            float bb = bias[col];
            #pragma unroll
            for (int v = 0; v < 8; ++v) {
                float gv = acc[t][v] + bb;
                gv = 0.5f * gv * (1.0f + erff(gv * 0.70710678118654752f));
                o[(size_t)(rowBase + rb + v) * Nc + col] = (_Float16)gv;
            }
        }
    }
}

// ---------------- flash attention per (b,h): online softmax, WMMA for QK^T and P.V ----------------
__global__ void attn_kernel(const _Float16* __restrict__ qkv, _Float16* __restrict__ o16,
                            int Nin, int Ntiles) {
    int b = blockIdx.x / HEADS;
    int h = blockIdx.x - b * HEADS;
    int l = threadIdx.x & 31;
    int w = threadIdx.x >> 5;
    int hi = l >> 4, ln = l & 15;
    __shared__ _Float16 pls[8][16 * 32];
    const _Float16* base = qkv + (size_t)b * NPAD * (3 * DIMC);
    const _Float16* qb = base + h * HDIM;
    const _Float16* kb = base + DIMC + h * HDIM;
    const _Float16* vb = base + 2 * DIMC + h * HDIM;
    int npairs = (Ntiles + 1) >> 1;
    for (int rt = w; rt < Ntiles; rt += 8) {
        // Q A-fragment (full head_dim = one K step)
        v16h aq;
        {
            const _Float16* p = qb + (size_t)(rt * 16 + ln) * (3 * DIMC) + hi * 16;
            #pragma unroll
            for (int e = 0; e < 8; ++e) { aq[e] = p[e]; aq[8 + e] = p[8 + e]; }
        }
        float m[8], ssum[8];
        #pragma unroll
        for (int v = 0; v < 8; ++v) { m[v] = NEGINF; ssum[v] = 0.0f; }
        v8f oc0 = {}, oc1 = {};
        for (int cp = 0; cp < npairs; ++cp) {
            int kc0 = cp * 32;
            int key0 = kc0 + ln;
            int key1 = key0 + 16;
            v16h bk0, bk1;
            {
                const _Float16* p0 = kb + (size_t)key0 * (3 * DIMC) + hi * 16;
                const _Float16* p1 = kb + (size_t)key1 * (3 * DIMC) + hi * 16;
                bool ok0 = key0 < Nin, ok1 = key1 < Nin;
                #pragma unroll
                for (int e = 0; e < 16; ++e) {
                    bk0[e] = ok0 ? p0[e] : (_Float16)0.0f;
                    bk1[e] = ok1 ? p1[e] : (_Float16)0.0f;
                }
            }
            v8f z = {};
            v8f s0 = __builtin_amdgcn_wmma_f32_16x16x32_f16(false, aq, false, bk0, (short)0, z, false, false);
            v8f s1 = __builtin_amdgcn_wmma_f32_16x16x32_f16(false, aq, false, bk1, (short)0, z, false, false);
            bool ok0 = key0 < Nin, ok1 = key1 < Nin;
            #pragma unroll
            for (int v = 0; v < 8; ++v) {
                float a0 = ok0 ? s0[v] * SCALE : NEGINF;
                float a1 = ok1 ? s1[v] * SCALE : NEGINF;
                float mx = hredMax(fmaxf(a0, a1));
                float mn = fmaxf(m[v], mx);
                float alpha = __expf(m[v] - mn);
                float p0 = __expf(a0 - mn);
                float p1 = __expf(a1 - mn);
                float ps = hredSum(p0 + p1);
                ssum[v] = ssum[v] * alpha + ps;
                oc0[v] *= alpha;
                oc1[v] *= alpha;
                m[v] = mn;
                int prow = hi * 8 + v;
                pls[w][prow * 32 + ln]      = (_Float16)p0;
                pls[w][prow * 32 + 16 + ln] = (_Float16)p1;
            }
            // P A-fragment from LDS (same-wave DS ops are in-order on CDNA5)
            v16h pa;
            {
                const _Float16* p = &pls[w][ln * 32];
                #pragma unroll
                for (int e = 0; e < 8; ++e) { pa[e] = p[hi * 16 + e]; pa[8 + e] = p[hi * 16 + 8 + e]; }
            }
            // V B-fragments (32 keys x 32 dims -> two 16-col halves)
            v16h bv0, bv1;
            #pragma unroll
            for (int e = 0; e < 16; ++e) {
                int key = kc0 + hi * 16 + e;
                bool ok = key < Nin;
                const _Float16* p = vb + (size_t)key * (3 * DIMC);
                bv0[e] = ok ? p[ln]      : (_Float16)0.0f;
                bv1[e] = ok ? p[16 + ln] : (_Float16)0.0f;
            }
            oc0 = __builtin_amdgcn_wmma_f32_16x16x32_f16(false, pa, false, bv0, (short)0, oc0, false, false);
            oc1 = __builtin_amdgcn_wmma_f32_16x16x32_f16(false, pa, false, bv1, (short)0, oc1, false, false);
        }
        #pragma unroll
        for (int v = 0; v < 8; ++v) {
            float inv = 1.0f / ssum[v];
            int row = rt * 16 + hi * 8 + v;
            size_t ob = (size_t)(b * NPAD + row) * DIMC + h * HDIM;
            o16[ob + ln]      = (_Float16)(oc0[v] * inv);
            o16[ob + 16 + ln] = (_Float16)(oc1[v] * inv);
        }
    }
}

// ---------------- pruning: human attention score over keypoint rows ----------------
__global__ void zero_human(float* __restrict__ human) {
    human[blockIdx.x * 256 + threadIdx.x] = 0.0f;
}
__global__ void human_kernel(const _Float16* __restrict__ qkv, float* __restrict__ human, int Nin) {
    int b = blockIdx.x / HEADS;
    int h = blockIdx.x - b * HEADS;
    int tid = threadIdx.x;
    const _Float16* base = qkv + (size_t)b * NPAD * (3 * DIMC);
    __shared__ float srow[NPAD + 32];
    __shared__ float red[256];
    for (int i = 0; i < JTOK; ++i) {
        const _Float16* qp = base + (size_t)i * (3 * DIMC) + h * HDIM;
        for (int key = tid; key < Nin; key += 256) {
            const _Float16* kp = base + (size_t)key * (3 * DIMC) + DIMC + h * HDIM;
            float acc = 0.0f;
            #pragma unroll
            for (int d = 0; d < HDIM; ++d) acc += (float)qp[d] * (float)kp[d];
            srow[key] = acc * SCALE;
        }
        __syncthreads();
        float lm = NEGINF;
        for (int key = tid; key < Nin; key += 256) lm = fmaxf(lm, srow[key]);
        red[tid] = lm; __syncthreads();
        for (int s = 128; s > 0; s >>= 1) { if (tid < s) red[tid] = fmaxf(red[tid], red[tid + s]); __syncthreads(); }
        float mrow = red[0]; __syncthreads();
        float ls = 0.0f;
        for (int key = tid; key < Nin; key += 256) ls += __expf(srow[key] - mrow);
        red[tid] = ls; __syncthreads();
        for (int s = 128; s > 0; s >>= 1) { if (tid < s) red[tid] += red[tid + s]; __syncthreads(); }
        float inv = 1.0f / (red[0] * (float)HEADS); __syncthreads();
        for (int key = tid; key < Nin; key += 256)
            if (key >= JTOK)
                atomicAdd(&human[b * 256 + key - JTOK], __expf(srow[key] - mrow) * inv);
        __syncthreads();
    }
}
// top-k (descending, ties -> lowest index first, matching jax.lax.top_k)
__global__ void topk_kernel(const float* __restrict__ human, int* __restrict__ idxout,
                            int Pin, int keep) {
    int b = blockIdx.x, tid = threadIdx.x;
    __shared__ float vals[256];
    __shared__ float rv[256];
    __shared__ int   ri[256];
    vals[tid] = (tid < Pin) ? human[b * 256 + tid] : -3.0e38f;
    __syncthreads();
    for (int t = 0; t < keep; ++t) {
        rv[tid] = vals[tid]; ri[tid] = tid; __syncthreads();
        for (int s = 128; s > 0; s >>= 1) {
            if (tid < s) {
                float a = rv[tid], c = rv[tid + s];
                int ia = ri[tid], ic = ri[tid + s];
                if (c > a || (c == a && ic < ia)) { rv[tid] = c; ri[tid] = ic; }
            }
            __syncthreads();
        }
        if (tid == 0) { idxout[b * 256 + t] = ri[0]; vals[ri[0]] = -3.0e38f; }
        __syncthreads();
    }
}
__global__ void gather_kernel(const float* __restrict__ xA, const float* __restrict__ posA,
                              float* __restrict__ xB, float* __restrict__ posB,
                              const int* __restrict__ idx, int keep) {
    int b = blockIdx.x, tid = threadIdx.x;
    for (int j = 0; j < JTOK; ++j)
        xB[((size_t)b * NPAD + j) * DIMC + tid] = xA[((size_t)b * NPAD + j) * DIMC + tid];
    for (int t = 0; t < keep; ++t) {
        int s = idx[b * 256 + t];
        xB[((size_t)b * NPAD + JTOK + t) * DIMC + tid] = xA[((size_t)b * NPAD + JTOK + s) * DIMC + tid];
        posB[((size_t)b * 256 + t) * DIMC + tid]       = posA[((size_t)b * 256 + s) * DIMC + tid];
    }
    for (int r = JTOK + keep; r < NPAD; ++r)
        xB[((size_t)b * NPAD + r) * DIMC + tid] = 0.0f;
}

// ---------------- final output: x[B,Nfin,256] then pos[B,Pfin,256], flat ----------------
__global__ void write_out_kernel(const float* __restrict__ xA, const float* __restrict__ posA,
                                 float* __restrict__ out, int Nfin, int Pfin) {
    size_t n1 = (size_t)BATCH * Nfin * DIMC;
    size_t tot = n1 + (size_t)BATCH * Pfin * DIMC;
    for (size_t i = blockIdx.x * blockDim.x + threadIdx.x; i < tot; i += (size_t)gridDim.x * blockDim.x) {
        if (i < n1) {
            int b = (int)(i / ((size_t)Nfin * DIMC));
            int r = (int)((i / DIMC) % Nfin);
            int d = (int)(i % DIMC);
            out[i] = xA[((size_t)b * NPAD + r) * DIMC + d];
        } else {
            size_t j = i - n1;
            int b = (int)(j / ((size_t)Pfin * DIMC));
            int r = (int)((j / DIMC) % Pfin);
            int d = (int)(j % DIMC);
            out[i] = posA[((size_t)b * 256 + r) * DIMC + d];
        }
    }
}

extern "C" void kernel_launch(void* const* d_in, const int* in_sizes, int n_in,
                              void* d_out, int out_size, void* d_ws, size_t ws_size,
                              hipStream_t stream) {
    (void)in_sizes; (void)n_in; (void)out_size; (void)ws_size;
    const float* x_in   = (const float*)d_in[0];
    const float* pos_in = (const float*)d_in[1];
    const float* ln1_g  = (const float*)d_in[2];
    const float* ln1_b  = (const float*)d_in[3];
    const float* w_qkv  = (const float*)d_in[4];
    const float* w_out  = (const float*)d_in[5];
    const float* b_out  = (const float*)d_in[6];
    const float* ln2_g  = (const float*)d_in[7];
    const float* ln2_b  = (const float*)d_in[8];
    const float* w1     = (const float*)d_in[9];
    const float* b1     = (const float*)d_in[10];
    const float* w2     = (const float*)d_in[11];
    const float* b2     = (const float*)d_in[12];

    size_t off = 0;
    char* ws = (char*)d_ws;
    auto carve = [&](size_t bytes) -> void* {
        void* p = ws + off;
        off += (bytes + 255) & ~(size_t)255;
        return p;
    };
    _Float16* wqkvT = (_Float16*)carve((size_t)12 * 256 * 768 * 2);
    _Float16* woutT = (_Float16*)carve((size_t)12 * 256 * 256 * 2);
    _Float16* w1T   = (_Float16*)carve((size_t)12 * 256 * 1024 * 2);
    _Float16* w2T   = (_Float16*)carve((size_t)12 * 1024 * 256 * 2);
    float*    xA    = (float*)carve((size_t)BATCH * NPAD * DIMC * 4);
    float*    xB    = (float*)carve((size_t)BATCH * NPAD * DIMC * 4);
    float*    posA  = (float*)carve((size_t)BATCH * 256 * DIMC * 4);
    float*    posB  = (float*)carve((size_t)BATCH * 256 * DIMC * 4);
    _Float16* xn16  = (_Float16*)carve((size_t)BATCH * NPAD * DIMC * 2);
    _Float16* qkv16 = (_Float16*)carve((size_t)BATCH * NPAD * 3 * DIMC * 2);
    _Float16* o16   = (_Float16*)carve((size_t)BATCH * NPAD * DIMC * 2);
    _Float16* h16   = (_Float16*)carve((size_t)BATCH * NPAD * MLPC * 2);
    float*    human = (float*)carve((size_t)BATCH * 256 * 4);
    int*      idxb  = (int*)carve((size_t)BATCH * 256 * 4);

    // weights -> f16 transposed [L][N][K]
    cvt_transpose<<<2048, 256, 0, stream>>>(w_qkv, wqkvT, 256, 768, 12 * 256 * 768);
    cvt_transpose<<<1024, 256, 0, stream>>>(w_out, woutT, 256, 256, 12 * 256 * 256);
    cvt_transpose<<<2048, 256, 0, stream>>>(w1, w1T, 256, 1024, 12 * 256 * 1024);
    cvt_transpose<<<2048, 256, 0, stream>>>(w2, w2T, 1024, 256, 12 * 1024 * 256);

    init_x<<<2048, 256, 0, stream>>>(x_in, xA, BATCH * NPAD * DIMC);
    init_pos<<<2048, 256, 0, stream>>>(pos_in, posA, BATCH * 256 * DIMC);

    int P = 256;
    for (int i = 0; i < 12; ++i) {
        int Nin = JTOK + P;
        if (i > 0) add_pos_kernel<<<BATCH * P, 256, 0, stream>>>(xA, posA, P);

        // --- attention block ---
        ln_kernel<<<BATCH * NPAD, 256, 0, stream>>>(xA, ln1_g + i * 256, ln1_b + i * 256, xn16, Nin);
        int tpb = (Nin + 15) / 16;
        int TM = BATCH * tpb;
        gemm_wmma<<<dim3(768 / 64, (TM + 7) / 8), 256, 0, stream>>>(
            xn16, wqkvT + (size_t)i * 256 * 768, nullptr, nullptr, qkv16, 256, 768, tpb, TM, 0);
        attn_kernel<<<BATCH * HEADS, 256, 0, stream>>>(qkv16, o16, Nin, tpb);
        gemm_wmma<<<dim3(256 / 64, (TM + 7) / 8), 256, 0, stream>>>(
            o16, woutT + (size_t)i * 256 * 256, b_out + i * 256, xA, xA, 256, 256, tpb, TM, 1);

        // --- token pruning ---
        if (i == 3 || i == 6 || i == 9) {
            zero_human<<<BATCH, 256, 0, stream>>>(human);
            human_kernel<<<BATCH * HEADS, 256, 0, stream>>>(qkv16, human, Nin);
            int keep = (P * 7 + 9) / 10;  // ceil(P * 0.7): 256->180->126->89
            topk_kernel<<<BATCH, 256, 0, stream>>>(human, idxb, P, keep);
            gather_kernel<<<BATCH, 256, 0, stream>>>(xA, posA, xB, posB, idxb, keep);
            float* t;
            t = xA; xA = xB; xB = t;
            t = posA; posA = posB; posB = t;
            P = keep;
            Nin = JTOK + P;
            tpb = (Nin + 15) / 16;
            TM = BATCH * tpb;
        }

        // --- FFN block ---
        ln_kernel<<<BATCH * NPAD, 256, 0, stream>>>(xA, ln2_g + i * 256, ln2_b + i * 256, xn16, Nin);
        gemm_wmma<<<dim3(1024 / 64, (TM + 7) / 8), 256, 0, stream>>>(
            xn16, w1T + (size_t)i * 256 * 1024, b1 + i * 1024, nullptr, h16, 256, 1024, tpb, TM, 2);
        gemm_wmma<<<dim3(256 / 64, (TM + 7) / 8), 256, 0, stream>>>(
            h16, w2T + (size_t)i * 1024 * 256, b2 + i * 256, xA, xA, 1024, 256, tpb, TM, 1);
    }

    write_out_kernel<<<4096, 256, 0, stream>>>(xA, posA, (float*)d_out, JTOK + P, P);
}
